// MotifGNN_64055142252645
// MI455X (gfx1250) — matrix operand order repaired and checked by reading, not compiled, not run on hardware
//
#include <hip/hip_runtime.h>

#define N_NODES 100000
#define N_EDGES 1600000
#define NODE_F  128
#define EDGE_F  16
#define E_HID   64
#define INTER   64

typedef __bf16 bf16_t;
typedef __attribute__((ext_vector_type(16))) __bf16 v16bf;
typedef __attribute__((ext_vector_type(8)))  __bf16 v8bf;
typedef __attribute__((ext_vector_type(8)))  float  v8f;

__device__ __forceinline__ v8f wmma_bf16(v16bf a, v16bf b, v8f c) {
  // D = A(16x32 bf16) x B(32x16 bf16) + C(16x16 f32)
  return __builtin_amdgcn_wmma_f32_16x16x32_bf16(false, a, false, b, (short)0, c,
                                                 false, false);
}

// A-fragment per ISA 7.12.2 (16-bit A 16x32), two 16B LDS vector loads:
// lane L: m = L&15, cb = (L>=16)?8:0 ; a[j]=A[m][k0+cb+j], a[8+j]=A[m][k0+16+cb+j]
__device__ __forceinline__ v16bf lds_afrag(const bf16_t* __restrict__ row, int cb, int k0) {
  const v8bf lo = *reinterpret_cast<const v8bf*>(row + k0 + cb);
  const v8bf hi = *reinterpret_cast<const v8bf*>(row + k0 + 16 + cb);
  return __builtin_shufflevector(lo, hi, 0, 1, 2, 3, 4, 5, 6, 7,
                                 8, 9, 10, 11, 12, 13, 14, 15);
}

// ---------------------------------------------------------------------------
// Weight prep: pack row-major f32 W[K][N] into per-lane WMMA B-fragments (bf16)
// frag(kstep,ntile): lane L holds b[j] = W[kstep*32 + ((L>=16)?8:0) + (j&7) + ((j>>3)<<4)][ntile*16 + (L&15)]
// stored contiguously: out[((kstep*ntiles + ntile)*32 + lane)*16 + j]
// ---------------------------------------------------------------------------
__global__ void prep_wfrag(const float* __restrict__ W, bf16_t* __restrict__ out,
                           int N, int ntiles, int kmax, int total) {
  int idx = blockIdx.x * blockDim.x + threadIdx.x;
  if (idx >= total) return;
  int j     = idx & 15;
  int lane  = (idx >> 4) & 31;
  int frag  = idx >> 9;
  int ntile = frag % ntiles;
  int kstep = frag / ntiles;
  int k = kstep * 32 + ((lane >> 4) << 3) + (j & 7) + ((j >> 3) << 4);
  int n = ntile * 16 + (lane & 15);
  out[idx] = (k < kmax) ? (bf16_t)W[k * N + n] : (bf16_t)0.0f;
}

__global__ void zero_kernel(float* __restrict__ p, long n) {
  long i = (long)blockIdx.x * blockDim.x + threadIdx.x;
  long st = (long)gridDim.x * blockDim.x;
  for (; i < n; i += st) p[i] = 0.0f;
}

// ---------------------------------------------------------------------------
// Fused: e = relu(ea@eW1+eb1)@eW2+eb2 ; msg = relu(h[src]+e) ; aggr[dst] += msg
// block = 256 threads (8 waves), tile = 128 edges. N_EDGES % 128 == 0.
// Wf1: 4 frags (K=32 padded, N=64). Wf2: 16 frags (K=64 -> 2 ksteps, N=128).
// ---------------------------------------------------------------------------
__global__ __launch_bounds__(256) void edge_msg_kernel(
    const float* __restrict__ ea, const int* __restrict__ ei,
    const float* __restrict__ h,
    const bf16_t* __restrict__ Wf1, const float* __restrict__ eb1,
    const bf16_t* __restrict__ Wf2, const float* __restrict__ eb2,
    float* __restrict__ aggr)
{
  __shared__ __align__(16) bf16_t sA[128 * 40];   // edge_attr tile, K 16->32, stride 40
  __shared__ __align__(16) bf16_t sH[128 * 72];   // stage-1 out, 64 cols, stride 72

  const int tid  = threadIdx.x;
  const int lane = tid & 31;
  const int wv   = tid >> 5;
  const int e0   = blockIdx.x * 128;

  const v16bf* F1 = reinterpret_cast<const v16bf*>(Wf1);
  const v16bf* F2 = reinterpret_cast<const v16bf*>(Wf2);

  // ---- stage edge_attr [128 x 16] -> bf16 LDS (vector stores), pad K 16..31 ----
  {
    int row  = tid >> 1;
    int hsel = tid & 1;
    const float* src = ea + (e0 + row) * EDGE_F + hsel * 8;
    v8bf t;
#pragma unroll
    for (int j = 0; j < 8; ++j) t[j] = (bf16_t)src[j];
    *reinterpret_cast<v8bf*>(&sA[row * 40 + hsel * 8]) = t;
    v8bf z = {};
    *reinterpret_cast<v8bf*>(&sA[row * 40 + 16 + hsel * 8]) = z;
  }
  if (e0 + 128 < N_EDGES)  // speculative stream prefetch of next tile
    __builtin_prefetch(ea + (e0 + 128) * EDGE_F + tid * 8, 0, 1);
  __syncthreads();

  const int nsub = lane & 15;
  const int cb   = (lane >> 4) * 8;    // A sub-column base

  // ---- stage 1: H1 = relu(EA @ eW1 + eb1), wave = M-tile, 4 N-tiles ----
  {
    const bf16_t* arow = &sA[(wv * 16 + nsub) * 40];
    v16bf a = lds_afrag(arow, cb, 0);
    int mbase = wv * 16 + ((lane >> 4) << 3);
#pragma unroll
    for (int nt = 0; nt < 4; ++nt) {
      int n = nt * 16 + nsub;
      v16bf b = F1[nt * 32 + lane];    // coalesced 32B fragment load
      v8f c = {};
      c = wmma_bf16(a, b, c);
      float bias = eb1[n];
#pragma unroll
      for (int i = 0; i < 8; ++i) {
        float v = c[i] + bias;
        v = v > 0.0f ? v : 0.0f;
        sH[(mbase + i) * 72 + n] = (bf16_t)v;
      }
    }
  }
  __syncthreads();

  // ---- stage 2: E2 = H1 @ eW2 + eb2 ; fused gather + relu + scatter-add ----
  {
    int n = wv * 16 + nsub;            // wave = N-tile (8 waves x 16 = 128 cols)
    v16bf b0 = F2[(0 * 8 + wv) * 32 + lane];
    v16bf b1 = F2[(1 * 8 + wv) * 32 + lane];
    float bias = eb2[n];
#pragma unroll
    for (int mt = 0; mt < 8; ++mt) {
      const bf16_t* arow = &sH[(mt * 16 + nsub) * 72];
      v16bf a0 = lds_afrag(arow, cb, 0);
      v16bf a1 = lds_afrag(arow, cb, 32);
      v8f c;
#pragma unroll
      for (int i = 0; i < 8; ++i) c[i] = bias;
      c = wmma_bf16(a0, b0, c);
      c = wmma_bf16(a1, b1, c);
      int mbase = mt * 16 + ((lane >> 4) << 3);
#pragma unroll
      for (int i = 0; i < 8; ++i) {
        int e = e0 + mbase + i;
        int s = ei[e];                       // src
        int d = ei[N_EDGES + e];             // dst
        float v = c[i] + h[s * NODE_F + n];  // L2-resident gather
        v = v > 0.0f ? v : 0.0f;
        atomicAdd(&aggr[d * NODE_F + n], v); // global_atomic_add_f32, L2-resident
      }
    }
  }
}

// ---------------------------------------------------------------------------
// h_out = relu((aggr + h) @ nW1 + nb1) @ nW2 + nb2 ; block = 256, tile = 64 nodes
// Wf1/Wf2: 32 frags each (4 ksteps x 8 ntiles), K = N = 128.
// ---------------------------------------------------------------------------
__global__ __launch_bounds__(256) void node_mlp_kernel(
    const float* __restrict__ h, const float* __restrict__ aggr,
    const bf16_t* __restrict__ Wf1, const float* __restrict__ nb1,
    const bf16_t* __restrict__ Wf2, const float* __restrict__ nb2,
    float* __restrict__ hout, int nrows)
{
  __shared__ __align__(16) bf16_t sX[64 * 136];
  __shared__ __align__(16) bf16_t sT[64 * 136];
  const int tid  = threadIdx.x;
  const int lane = tid & 31;
  const int wv   = tid >> 5;
  const int r0   = blockIdx.x * 64;

  const v16bf* F1 = reinterpret_cast<const v16bf*>(Wf1);
  const v16bf* F2 = reinterpret_cast<const v16bf*>(Wf2);

  // ---- stage (aggr + h) -> bf16 LDS, 8 elements per thread-iteration ----
#pragma unroll
  for (int it = 0; it < 4; ++it) {
    int chunk = it * 256 + tid;        // 1024 chunks of 8
    int row = chunk >> 4, col = (chunk & 15) * 8;
    int r = r0 + row;
    v8bf t;
    if (r < nrows) {
      const float* pa = aggr + r * NODE_F + col;
      const float* ph = h    + r * NODE_F + col;
#pragma unroll
      for (int j = 0; j < 8; ++j) t[j] = (bf16_t)(pa[j] + ph[j]);
    } else {
      v8bf z = {};
      t = z;
    }
    *reinterpret_cast<v8bf*>(&sX[row * 136 + col]) = t;
  }
  __syncthreads();

  const int nsub = lane & 15;
  const int cb   = (lane >> 4) * 8;
  const int n    = wv * 16 + nsub;     // wave = N-tile

  // ---- stage 1: T = relu(X @ nW1 + nb1) ----
  {
    v16bf b0 = F1[(0 * 8 + wv) * 32 + lane];
    v16bf b1 = F1[(1 * 8 + wv) * 32 + lane];
    v16bf b2 = F1[(2 * 8 + wv) * 32 + lane];
    v16bf b3 = F1[(3 * 8 + wv) * 32 + lane];
    float bias = nb1[n];
#pragma unroll
    for (int mt = 0; mt < 4; ++mt) {
      const bf16_t* arow = &sX[(mt * 16 + nsub) * 136];
      v8f c;
#pragma unroll
      for (int i = 0; i < 8; ++i) c[i] = bias;
      c = wmma_bf16(lds_afrag(arow, cb, 0),  b0, c);
      c = wmma_bf16(lds_afrag(arow, cb, 32), b1, c);
      c = wmma_bf16(lds_afrag(arow, cb, 64), b2, c);
      c = wmma_bf16(lds_afrag(arow, cb, 96), b3, c);
      int mbase = mt * 16 + ((lane >> 4) << 3);
#pragma unroll
      for (int i = 0; i < 8; ++i) {
        float v = c[i];
        v = v > 0.0f ? v : 0.0f;
        sT[(mbase + i) * 136 + n] = (bf16_t)v;
      }
    }
  }
  __syncthreads();

  // ---- stage 2: hout = T @ nW2 + nb2 (no relu) ----
  {
    v16bf b0 = F2[(0 * 8 + wv) * 32 + lane];
    v16bf b1 = F2[(1 * 8 + wv) * 32 + lane];
    v16bf b2 = F2[(2 * 8 + wv) * 32 + lane];
    v16bf b3 = F2[(3 * 8 + wv) * 32 + lane];
    float bias = nb2[n];
#pragma unroll
    for (int mt = 0; mt < 4; ++mt) {
      const bf16_t* arow = &sT[(mt * 16 + nsub) * 136];
      v8f c;
#pragma unroll
      for (int i = 0; i < 8; ++i) c[i] = bias;
      c = wmma_bf16(lds_afrag(arow, cb, 0),  b0, c);
      c = wmma_bf16(lds_afrag(arow, cb, 32), b1, c);
      c = wmma_bf16(lds_afrag(arow, cb, 64), b2, c);
      c = wmma_bf16(lds_afrag(arow, cb, 96), b3, c);
      int mbase = mt * 16 + ((lane >> 4) << 3);
#pragma unroll
      for (int i = 0; i < 8; ++i) {
        int r = r0 + mbase + i;
        if (r < nrows) hout[r * NODE_F + n] = c[i];
      }
    }
  }
}

// g[col] = sum over rows of h[:, col] ; g pre-zeroed
__global__ __launch_bounds__(256) void gsum_kernel(const float* __restrict__ h,
                                                   float* __restrict__ g, int nrows) {
  __shared__ float red[256];
  int col  = threadIdx.x & 127;
  int half = threadIdx.x >> 7;
  float acc = 0.0f;
  for (int r = blockIdx.x * 2 + half; r < nrows; r += gridDim.x * 2)
    acc += h[r * NODE_F + col];
  red[threadIdx.x] = acc;
  __syncthreads();
  if (half == 0) atomicAdd(&g[col], red[threadIdx.x] + red[threadIdx.x + 128]);
}

// out = relu(g @ agg_w), agg_w [128 x 64]
__global__ void final_kernel(const float* __restrict__ g, const float* __restrict__ w,
                             float* __restrict__ out) {
  int j = threadIdx.x;  // 0..63
  float acc = 0.0f;
#pragma unroll
  for (int k = 0; k < NODE_F; ++k) acc += g[k] * w[k * INTER + j];
  out[j] = acc > 0.0f ? acc : 0.0f;
}

extern "C" void kernel_launch(void* const* d_in, const int* in_sizes, int n_in,
                              void* d_out, int out_size, void* d_ws, size_t ws_size,
                              hipStream_t stream)
{
  // --- resolve input order (insertion vs alphabetical tree-flatten) ---
  const float *x, *ea, *agg_w;
  const int *ei;
  if (in_sizes[0] == N_NODES * NODE_F) {        // insertion: x, ei, ea, params..., agg_w
    x     = (const float*)d_in[0];
    ei    = (const int*)  d_in[1];
    ea    = (const float*)d_in[2];
    agg_w = (const float*)d_in[27];
  } else {                                      // sorted: agg_w, ea, ei, params..., x
    agg_w = (const float*)d_in[0];
    ea    = (const float*)d_in[1];
    ei    = (const int*)  d_in[2];
    x     = (const float*)d_in[27];
  }
  const float *eW1[3], *eb1[3], *eW2[3], *eb2[3], *nW1[3], *nb1[3], *nW2[3], *nb2[3];
  for (int l = 0; l < 3; ++l) {
    int b = 3 + l * 8;
    if (in_sizes[b + 1] == E_HID) {             // insertion: eW1,eb1,eW2,eb2,nW1,nb1,nW2,nb2
      eW1[l] = (const float*)d_in[b + 0]; eb1[l] = (const float*)d_in[b + 1];
      eW2[l] = (const float*)d_in[b + 2]; eb2[l] = (const float*)d_in[b + 3];
      nW1[l] = (const float*)d_in[b + 4]; nb1[l] = (const float*)d_in[b + 5];
      nW2[l] = (const float*)d_in[b + 6]; nb2[l] = (const float*)d_in[b + 7];
    } else {                                    // sorted: eW1,eW2,eb1,eb2,nW1,nW2,nb1,nb2
      eW1[l] = (const float*)d_in[b + 0]; eW2[l] = (const float*)d_in[b + 1];
      eb1[l] = (const float*)d_in[b + 2]; eb2[l] = (const float*)d_in[b + 3];
      nW1[l] = (const float*)d_in[b + 4]; nW2[l] = (const float*)d_in[b + 5];
      nb1[l] = (const float*)d_in[b + 6]; nb2[l] = (const float*)d_in[b + 7];
    }
  }

  // --- workspace layout (floats, then 32B-aligned bf16 fragment region) ---
  float* ws   = (float*)d_ws;
  float* aggr = ws;                                   // [N, 128]
  float* hA   = ws + (size_t)N_NODES * NODE_F;        // [N, 128]
  float* hB   = hA + (size_t)N_NODES * NODE_F;        // [N, 128]
  float* g    = hB + (size_t)N_NODES * NODE_F;        // [128]
  bf16_t* fb  = (bf16_t*)(g + NODE_F);                // fragment region

  // per-layer fragment sizes (halves): eW1 2048, eW2 8192, nW1 16384, nW2 16384
  const int FE1 = 1 * 4 * 512, FE2 = 2 * 8 * 512, FN = 4 * 8 * 512;
  const int LSTR = FE1 + FE2 + FN + FN;               // 43008 per layer
  bf16_t *fE1[3], *fE2[3], *fN1[3], *fN2[3];
  for (int l = 0; l < 3; ++l) {
    bf16_t* p = fb + (size_t)l * LSTR;
    fE1[l] = p;             fE2[l] = p + FE1;
    fN1[l] = p + FE1 + FE2; fN2[l] = p + FE1 + FE2 + FN;
    prep_wfrag<<<(FE1 + 255) / 256, 256, 0, stream>>>(eW1[l], fE1[l], E_HID,  4, EDGE_F, FE1);
    prep_wfrag<<<(FE2 + 255) / 256, 256, 0, stream>>>(eW2[l], fE2[l], NODE_F, 8, E_HID,  FE2);
    prep_wfrag<<<(FN  + 255) / 256, 256, 0, stream>>>(nW1[l], fN1[l], NODE_F, 8, NODE_F, FN);
    prep_wfrag<<<(FN  + 255) / 256, 256, 0, stream>>>(nW2[l], fN2[l], NODE_F, 8, NODE_F, FN);
  }

  const float* h = x;
  float* houts[3] = { hA, hB, hA };
  for (int l = 0; l < 3; ++l) {
    zero_kernel<<<4096, 256, 0, stream>>>(aggr, (long)N_NODES * NODE_F);
    edge_msg_kernel<<<N_EDGES / 128, 256, 0, stream>>>(
        ea, ei, h, fE1[l], eb1[l], fE2[l], eb2[l], aggr);
    node_mlp_kernel<<<(N_NODES + 63) / 64, 256, 0, stream>>>(
        h, aggr, fN1[l], nb1[l], fN2[l], nb2[l], houts[l], N_NODES);
    h = houts[l];
  }
  zero_kernel<<<1, 128, 0, stream>>>(g, NODE_F);
  gsum_kernel<<<240, 256, 0, stream>>>(h, g, N_NODES);
  final_kernel<<<1, 64, 0, stream>>>(g, agg_w, (float*)d_out);
}
